// Model_58832462021324
// MI455X (gfx1250) — matrix operand order repaired and checked by reading, not compiled, not run on hardware
//
#include <hip/hip_runtime.h>
#include <hip/hip_bf16.h>

// ---------------------------------------------------------------------------
// Bipartite GAT (2 layers, H=4 heads, C=128) for MI455X / gfx1250, wave32.
//   - Hs = x_src @ W_src done with v_wmma_f32_16x16x32_bf16 (f32 accum).
//   - Hd / He GEMMs algebraically folded into tiny per-node / per-edge dots.
//   - segment softmax via float atomic max/add; scatter via atomic f32 adds.
//   - Layer-1 u2i conv is dead for the output xu and is skipped.
// ---------------------------------------------------------------------------

#define C_DIM   128
#define H_HEADS 4
#define CP_DIM  32
#define D_EDGE  16
#define GSTRIDE 130   // LDS row stride (bf16 elems) for staged weights, bank skew

typedef __attribute__((ext_vector_type(16))) __bf16 v16bf;
typedef __attribute__((ext_vector_type(8)))  float  v8f;

__device__ __forceinline__ unsigned pack_bf16_rne(float a, float b) {
    unsigned ua = __float_as_uint(a);
    unsigned ub = __float_as_uint(b);
    ua = (ua + 0x7FFFu + ((ua >> 16) & 1u)) >> 16;   // round-to-nearest-even
    ub = (ub + 0x7FFFu + ((ub >> 16) & 1u)) >> 16;
    return (ua & 0xFFFFu) | (ub << 16);
}

__device__ __forceinline__ void atomicMaxF(float* addr, float val) {
    // Monotone in float order for arbitrary sign mixes; init must be -inf.
    if (val >= 0.0f) atomicMax((int*)addr, __float_as_int(val));
    else             atomicMin((unsigned int*)addr, __float_as_uint(val));
}

// ---------------------------------------------------------------------------
// GEMM: H[N,128] = bf16(X[N,128]) @ bf16(W[128,128]), f32 accumulate (WMMA).
// Block = 256 threads = 8 waves; wave w owns rows [blk*128 + 16w, +16),
// all 8 column tiles. W staged to LDS transposed (Wt[n][k]) so the two
// consecutive-K bf16 of a B fragment are one aligned 32-bit LDS read.
// ---------------------------------------------------------------------------
__global__ __launch_bounds__(256) void gemm128_wmma(
    const float* __restrict__ X, const float* __restrict__ W,
    float* __restrict__ Hout, int Nrows) {
    __shared__ unsigned short Wt[C_DIM * GSTRIDE];

    const int tid = threadIdx.x;
    for (int i = tid; i < C_DIM * C_DIM; i += 256) {
        int k = i >> 7, n = i & 127;
        unsigned uw = __float_as_uint(W[i]);
        uw = (uw + 0x7FFFu + ((uw >> 16) & 1u)) >> 16;
        Wt[n * GSTRIDE + k] = (unsigned short)uw;
    }
    __syncthreads();

    const int wave = tid >> 5, lane = tid & 31;
    const int m = lane & 15, half = lane >> 4;
    const int rowBase = blockIdx.x * 128 + wave * 16;
    int rowA = rowBase + m;
    if (rowA >= Nrows) rowA = Nrows - 1;            // clamp loads; stores guarded
    const float* arow = X + (size_t)rowA * C_DIM;

    v8f acc[8] = {};

    for (int kc = 0; kc < C_DIM; kc += 32) {
        // A fragment: 16-bit A 16x32 layout (ISA 7.12.2):
        //   v<4 : K = kc + 2v + 8*half  (pairs)
        //   v>=4: K = kc + 16 + 2(v-4) + 8*half
        union { unsigned u[8]; v16bf v; } afrag;
#pragma unroll
        for (int vv = 0; vv < 8; ++vv) {
            int kb = kc + ((vv < 4) ? (2 * vv + 8 * half)
                                    : (16 + 2 * (vv - 4) + 8 * half));
            float2 f = *(const float2*)(arow + kb);
            afrag.u[vv] = pack_bf16_rne(f.x, f.y);
        }
#pragma unroll
        for (int t = 0; t < 8; ++t) {
            // B fragment: lane n = t*16 + m, K pairs same as A fragment
            union { unsigned u[8]; v16bf v; } bfrag;
            const int n = t * 16 + m;
#pragma unroll
            for (int vv = 0; vv < 8; ++vv) {
                int kb = kc + ((vv < 4) ? (2 * vv + 8 * half)
                                        : (16 + 2 * (vv - 4) + 8 * half));
                bfrag.u[vv] =
                    *(const unsigned*)((const unsigned short*)Wt + n * GSTRIDE + kb);
            }
            acc[t] = __builtin_amdgcn_wmma_f32_16x16x32_bf16(
                false, afrag.v, false, bfrag.v, (short)0, acc[t], false, false);
        }
    }

    // D layout: lanes 0-15 -> M = i, lanes 16-31 -> M = 8+i; N = t*16 + m.
#pragma unroll
    for (int t = 0; t < 8; ++t) {
#pragma unroll
        for (int i = 0; i < 8; ++i) {
            int r = rowBase + i + 8 * half;
            if (r < Nrows) Hout[(size_t)r * C_DIM + t * 16 + m] = acc[t][i];
        }
    }
}

// Fold Wd*a_d -> Vd[128][4] and We*a_e -> Ve[4][16] (single tiny block).
__global__ void precompute_folds(const float* __restrict__ Wd,
                                 const float* __restrict__ ad,
                                 const float* __restrict__ We,
                                 const float* __restrict__ ae,
                                 float* __restrict__ Vd, float* __restrict__ Ve) {
    for (int idx = threadIdx.x; idx < 512 + 64; idx += blockDim.x) {
        if (idx < 512) {
            int c = idx >> 2, h = idx & 3;
            float s = 0.f;
            for (int j = 0; j < CP_DIM; ++j)
                s += Wd[c * C_DIM + h * CP_DIM + j] * ad[h * CP_DIM + j];
            Vd[c * 4 + h] = s;
        } else {
            int k = idx - 512, h = k >> 4, d = k & 15;
            float s = 0.f;
            for (int j = 0; j < CP_DIM; ++j)
                s += We[d * C_DIM + h * CP_DIM + j] * ae[h * CP_DIM + j];
            Ve[h * D_EDGE + d] = s;
        }
    }
}

// es[n,h] = sum_c Hs[n, h*32+c] * a_s[h,c]  (wave per node)
__global__ __launch_bounds__(256) void node_scores(
    const float* __restrict__ H, const float* __restrict__ a,
    float* __restrict__ es, int N) {
    int wid = (int)((blockIdx.x * 256 + threadIdx.x) >> 5);
    int lane = threadIdx.x & 31;
    if (wid >= N) return;
    float4 h4 = *(const float4*)(H + (size_t)wid * C_DIM + lane * 4);
    float4 a4 = *(const float4*)(a + lane * 4);   // a flattened [4][32] == col index
    float p = h4.x * a4.x + h4.y * a4.y + h4.z * a4.z + h4.w * a4.w;
    p += __shfl_xor(p, 1, 32);
    p += __shfl_xor(p, 2, 32);
    p += __shfl_xor(p, 4, 32);
    if ((lane & 7) == 0) es[(size_t)wid * 4 + (lane >> 3)] = p;
}

// ed[n,h] = x_dst[n,:] . Vd[:,h]  (wave per node)
__global__ __launch_bounds__(256) void dst_scores(
    const float* __restrict__ X, const float* __restrict__ Vd,
    float* __restrict__ ed, int N) {
    int wid = (int)((blockIdx.x * 256 + threadIdx.x) >> 5);
    int lane = threadIdx.x & 31;
    if (wid >= N) return;
    float4 x4 = *(const float4*)(X + (size_t)wid * C_DIM + lane * 4);
    float a0 = 0.f, a1 = 0.f, a2 = 0.f, a3 = 0.f;
#pragma unroll
    for (int j = 0; j < 4; ++j) {
        float xv = (&x4.x)[j];
        float4 v = *(const float4*)(Vd + (lane * 4 + j) * 4);
        a0 += xv * v.x; a1 += xv * v.y; a2 += xv * v.z; a3 += xv * v.w;
    }
    for (int msk = 1; msk < 32; msk <<= 1) {
        a0 += __shfl_xor(a0, msk, 32);
        a1 += __shfl_xor(a1, msk, 32);
        a2 += __shfl_xor(a2, msk, 32);
        a3 += __shfl_xor(a3, msk, 32);
    }
    if (lane == 0) {
        float4 r; r.x = a0; r.y = a1; r.z = a2; r.w = a3;
        *(float4*)(ed + (size_t)wid * 4) = r;
    }
}

__global__ void fill_neg_inf(float* __restrict__ p, int n) {
    int i = blockIdx.x * blockDim.x + threadIdx.x;
    if (i < n) p[i] = __int_as_float(0xFF800000);
}

// alpha[e,h] = leaky_relu(es[src]+ed[dst]+ea[e].Ve[h]); atomic segment max.
__global__ __launch_bounds__(256) void edge_alpha_max(
    const int* __restrict__ src, const int* __restrict__ dst,
    const float* __restrict__ ea, const float* __restrict__ Ve,
    const float* __restrict__ es, const float* __restrict__ ed,
    float* __restrict__ alpha, float* __restrict__ mx, int E) {
    int e = blockIdx.x * 256 + threadIdx.x;
    if (e >= E) return;
    int s = src[e], d = dst[e];
    float4 es4 = *(const float4*)(es + (size_t)s * 4);
    float4 ed4 = *(const float4*)(ed + (size_t)d * 4);
    float ar[D_EDGE];
#pragma unroll
    for (int j = 0; j < D_EDGE; ++j) ar[j] = ea[(size_t)e * D_EDGE + j];
#pragma unroll
    for (int h = 0; h < H_HEADS; ++h) {
        float ee = 0.f;
#pragma unroll
        for (int j = 0; j < D_EDGE; ++j) ee += ar[j] * Ve[h * D_EDGE + j];
        float a = (&es4.x)[h] + (&ed4.x)[h] + ee;
        a = a > 0.f ? a : 0.2f * a;               // NEG_SLOPE = 0.2
        alpha[(size_t)e * 4 + h] = a;
        atomicMaxF(mx + (size_t)d * 4 + h, a);
    }
}

// ex = exp(alpha - mx[dst]); den[dst] += ex  (ex overwrites alpha buffer)
__global__ __launch_bounds__(256) void edge_exp_den(
    const int* __restrict__ dst, const float* __restrict__ mx,
    float* __restrict__ alpha, float* __restrict__ den, int E) {
    int e = blockIdx.x * 256 + threadIdx.x;
    if (e >= E) return;
    int d = dst[e];
    float4 a4 = *(float4*)(alpha + (size_t)e * 4);
    float4 m4 = *(const float4*)(mx + (size_t)d * 4);
    float4 x4;
    x4.x = __expf(a4.x - m4.x);
    x4.y = __expf(a4.y - m4.y);
    x4.z = __expf(a4.z - m4.z);
    x4.w = __expf(a4.w - m4.w);
    *(float4*)(alpha + (size_t)e * 4) = x4;
    atomicAdd(den + (size_t)d * 4 + 0, x4.x);
    atomicAdd(den + (size_t)d * 4 + 1, x4.y);
    atomicAdd(den + (size_t)d * 4 + 2, x4.z);
    atomicAdd(den + (size_t)d * 4 + 3, x4.w);
}

// out[dst,:] += Hs[src,:] * w   (wave per edge; lane owns 4 contiguous cols)
__global__ __launch_bounds__(256) void edge_scatter(
    const int* __restrict__ src, const int* __restrict__ dst,
    const float* __restrict__ Hs, const float* __restrict__ ex,
    const float* __restrict__ den, float* __restrict__ out, int E) {
    int e = (int)((blockIdx.x * 256 + threadIdx.x) >> 5);
    int lane = threadIdx.x & 31;
    if (e >= E) return;
    int s = src[e], d = dst[e];
    int h = lane >> 3;                            // 4 cols per lane, 32 per head
    float w = ex[(size_t)e * 4 + h] / (den[(size_t)d * 4 + h] + 1e-16f);
    float4 h4 = *(const float4*)(Hs + (size_t)s * C_DIM + lane * 4);
    float* orow = out + (size_t)d * C_DIM + lane * 4;
    atomicAdd(orow + 0, h4.x * w);
    atomicAdd(orow + 1, h4.y * w);
    atomicAdd(orow + 2, h4.z * w);
    atomicAdd(orow + 3, h4.w * w);
}

// y = relu(LN(acc + bias) * gamma + beta)   (wave per row)
__global__ __launch_bounds__(256) void ln_relu(
    const float* __restrict__ acc, const float* __restrict__ bias,
    const float* __restrict__ gamma, const float* __restrict__ beta,
    float* __restrict__ out, int N) {
    int n = (int)((blockIdx.x * 256 + threadIdx.x) >> 5);
    int lane = threadIdx.x & 31;
    if (n >= N) return;
    float4 v = *(const float4*)(acc + (size_t)n * C_DIM + lane * 4);
    float4 b4 = *(const float4*)(bias + lane * 4);
    v.x += b4.x; v.y += b4.y; v.z += b4.z; v.w += b4.w;
    float s = v.x + v.y + v.z + v.w;
    float q = v.x * v.x + v.y * v.y + v.z * v.z + v.w * v.w;
    for (int msk = 1; msk < 32; msk <<= 1) {
        s += __shfl_xor(s, msk, 32);
        q += __shfl_xor(q, msk, 32);
    }
    float mu  = s * (1.0f / 128.0f);
    float var = q * (1.0f / 128.0f) - mu * mu;
    float inv = rsqrtf(var + 1e-5f);
    float4 g4 = *(const float4*)(gamma + lane * 4);
    float4 e4 = *(const float4*)(beta + lane * 4);
    float4 r;
    r.x = fmaxf((v.x - mu) * inv * g4.x + e4.x, 0.f);
    r.y = fmaxf((v.y - mu) * inv * g4.y + e4.y, 0.f);
    r.z = fmaxf((v.z - mu) * inv * g4.z + e4.z, 0.f);
    r.w = fmaxf((v.w - mu) * inv * g4.w + e4.w, 0.f);
    *(float4*)(out + (size_t)n * C_DIM + lane * 4) = r;
}

// ---------------------------------------------------------------------------
extern "C" void kernel_launch(void* const* d_in, const int* in_sizes, int n_in,
                              void* d_out, int out_size, void* d_ws, size_t ws_size,
                              hipStream_t stream) {
    const float* x_user  = (const float*)d_in[0];
    const float* x_item  = (const float*)d_in[1];
    const int*   ei_u2i  = (const int*)d_in[2];
    const int*   ei_i2u  = (const int*)d_in[3];
    const float* ea_u2i  = (const float*)d_in[4];
    const float* ea_i2u  = (const float*)d_in[5];
    const float* W_src   = (const float*)d_in[6];
    const float* W_dst   = (const float*)d_in[7];
    const float* W_edge  = (const float*)d_in[8];
    const float* att_src = (const float*)d_in[9];
    const float* att_dst = (const float*)d_in[10];
    const float* att_edge= (const float*)d_in[11];
    const float* bias    = (const float*)d_in[12];
    const float* ln_g    = (const float*)d_in[13];
    const float* ln_b    = (const float*)d_in[14];

    const int NU = in_sizes[0] / C_DIM;
    const int NI = in_sizes[1] / C_DIM;
    const int E  = in_sizes[2] / 2;
    const int NMAX = (NU > NI) ? NU : NI;

    // -- workspace carve-out ------------------------------------------------
    char* ws = (char*)d_ws;
    size_t off = 0;
    auto carve = [&](size_t bytes) -> float* {
        float* p = (float*)(ws + off);
        off += (bytes + 255) & ~(size_t)255;
        return p;
    };
    float* Hs   = carve((size_t)NMAX * C_DIM * 4);
    float* xu_n = carve((size_t)NU   * C_DIM * 4);
    float* xi_n = carve((size_t)NI   * C_DIM * 4);
    float* accb = carve((size_t)NMAX * C_DIM * 4);
    float* es   = carve((size_t)NMAX * 4 * 4);
    float* ed   = carve((size_t)NMAX * 4 * 4);
    float* alpha= carve((size_t)E * 4 * 4);
    float* mx   = carve((size_t)NMAX * 4 * 4);
    float* den  = carve((size_t)NMAX * 4 * 4);
    float* Vd   = carve(512 * 4);
    float* Ve   = carve(64 * 4);

    const int C2 = C_DIM * C_DIM, DC = D_EDGE * C_DIM, A2 = H_HEADS * CP_DIM,
              B2 = C_DIM;

    auto run_conv = [&](const float* Xs, const float* Xd, const int* srcI,
                        const int* dstI, const float* ea, int Ns, int Nd,
                        const float* Wsrc, const float* Wdst, const float* Wedg,
                        const float* as_, const float* ad_, const float* ae_,
                        const float* b_, const float* g_, const float* be_,
                        float* xout) {
        gemm128_wmma<<<(Ns + 127) / 128, 256, 0, stream>>>(Xs, Wsrc, Hs, Ns);
        precompute_folds<<<1, 256, 0, stream>>>(Wdst, ad_, Wedg, ae_, Vd, Ve);
        node_scores<<<(Ns + 7) / 8, 256, 0, stream>>>(Hs, as_, es, Ns);
        dst_scores<<<(Nd + 7) / 8, 256, 0, stream>>>(Xd, Vd, ed, Nd);
        hipMemsetAsync(den, 0, (size_t)Nd * 4 * 4, stream);
        hipMemsetAsync(accb, 0, (size_t)Nd * C_DIM * 4, stream);
        fill_neg_inf<<<(Nd * 4 + 255) / 256, 256, 0, stream>>>(mx, Nd * 4);
        edge_alpha_max<<<(E + 255) / 256, 256, 0, stream>>>(srcI, dstI, ea, Ve,
                                                            es, ed, alpha, mx, E);
        edge_exp_den<<<(E + 255) / 256, 256, 0, stream>>>(dstI, mx, alpha, den, E);
        edge_scatter<<<(E + 7) / 8, 256, 0, stream>>>(srcI, dstI, Hs, alpha, den,
                                                      accb, E);
        ln_relu<<<(Nd + 7) / 8, 256, 0, stream>>>(accb, b_, g_, be_, xout, Nd);
    };

    // layer 0, user->item (params [0,0]; LN item = [0,1])
    run_conv(x_user, x_item, ei_u2i, ei_u2i + E, ea_u2i, NU, NI,
             W_src + 0 * C2, W_dst + 0 * C2, W_edge + 0 * DC,
             att_src + 0 * A2, att_dst + 0 * A2, att_edge + 0 * A2,
             bias + 0 * B2, ln_g + 1 * B2, ln_b + 1 * B2, xi_n);
    // layer 0, item->user (params [0,1]; LN user = [0,0])
    run_conv(x_item, x_user, ei_i2u, ei_i2u + E, ea_i2u, NI, NU,
             W_src + 1 * C2, W_dst + 1 * C2, W_edge + 1 * DC,
             att_src + 1 * A2, att_dst + 1 * A2, att_edge + 1 * A2,
             bias + 1 * B2, ln_g + 0 * B2, ln_b + 0 * B2, xu_n);
    // layer 1, item->user only (params [1,1]; LN user = [1,0]); u2i is dead
    // for the returned xu. Final LN+ReLU writes straight into d_out.
    run_conv(xi_n, xu_n, ei_i2u, ei_i2u + E, ea_i2u, NI, NU,
             W_src + 3 * C2, W_dst + 3 * C2, W_edge + 3 * DC,
             att_src + 3 * A2, att_dst + 3 * A2, att_edge + 3 * A2,
             bias + 3 * B2, ln_g + 2 * B2, ln_b + 2 * B2, (float*)d_out);
}
